// AttnFree_Block_48455821034243
// MI455X (gfx1250) — matrix operand order repaired and checked by reading, not compiled, not run on hardware
//
#include <hip/hip_runtime.h>
#include <hip/hip_bf16.h>

// AttnFree transformer block for gfx1250 (MI455X), wave32 WMMA path.
// All GEMMs lowered to v_wmma_f32_16x16x32_bf16 (f32 accumulate), 16x64
// output tile per wave (4 accumulators reuse one A fragment), explicit
// double-buffered global->VGPR fragment pipeline.
// DistAngleFusion folded algebraically: O(N^2 D^2) -> O(N^2 D).

#define BB 2
#define NN 512
#define DD 128
#define FH 512
#define MM (BB * NN)   // 1024 rows

typedef __attribute__((ext_vector_type(16))) __bf16 v16bf;
typedef __attribute__((ext_vector_type(8)))  float  v8f;

__device__ __forceinline__ unsigned short f2bf(float x) {
    unsigned int u = __float_as_uint(x);
    unsigned int r = (u + 0x7FFFu + ((u >> 16) & 1u)) >> 16;
    return (unsigned short)r;
}
__device__ __forceinline__ float frcp(float x) { return __builtin_amdgcn_rcpf(x); }
__device__ __forceinline__ float frsq(float x) { return __builtin_amdgcn_rsqf(x); }
__device__ __forceinline__ float bsig(float x) { return frcp(1.0f + __expf(-x)); }

// ---------------------------------------------------------------------------
// Weight pack: (K,O) f32 row-major -> (O,K) bf16 row-major (transposed)
// so that a lane's (k,k+1) bf16 pair for a fixed output column is one dword.
// ---------------------------------------------------------------------------
__global__ void packwt_kernel(const float* __restrict__ W, unsigned short* __restrict__ Wt,
                              int K, int O) {
    int idx = blockIdx.x * 256 + threadIdx.x;
    if (idx >= K * O) return;
    int k = idx / O, o = idx % O;
    Wt[o * K + k] = f2bf(W[idx]);
}

// ---------------------------------------------------------------------------
// Fold DistAngleFusion: u_d = W2d@outlin_w, v_d = W2d@gate_w[:D], etc.
// fold layout: [u_d(128), v_d(128), u_a(128), v_a(128), c_d, e_d, c_a, e_a]
// ---------------------------------------------------------------------------
__global__ void fold_kernel(const float* __restrict__ dw2, const float* __restrict__ db2,
                            const float* __restrict__ aw2, const float* __restrict__ ab2,
                            const float* __restrict__ outw, const float* __restrict__ gatew,
                            float* __restrict__ fold) {
    int j = threadIdx.x;  // 128 threads
    float ud = 0.f, vd = 0.f, ua = 0.f, va = 0.f;
    for (int o = 0; o < DD; ++o) {
        float ow = outw[o], gd = gatew[o], ga = gatew[DD + o];
        float d2 = dw2[j * DD + o], a2 = aw2[j * DD + o];
        ud += d2 * ow; vd += d2 * gd;
        ua += a2 * ow; va += a2 * ga;
    }
    fold[j] = ud; fold[128 + j] = vd; fold[256 + j] = ua; fold[384 + j] = va;
    if (j < 4) {
        float s = 0.f;
        for (int o = 0; o < DD; ++o) {
            if      (j == 0) s += db2[o] * outw[o];
            else if (j == 1) s += db2[o] * gatew[o];
            else if (j == 2) s += ab2[o] * outw[o];
            else             s += ab2[o] * gatew[DD + o];
        }
        fold[512 + j] = s;
    }
}

// ---------------------------------------------------------------------------
// Instance norm over tokens (axis=1), per (batch, feature); biased variance.
// Optional residual input R; writes f32 and/or bf16.
// grid (D/32, B), block 256 (32 features x 8 token strips)
// ---------------------------------------------------------------------------
__global__ void inorm_kernel(const float* __restrict__ X, const float* __restrict__ R,
                             const float* __restrict__ w, const float* __restrict__ b,
                             float* __restrict__ Yf, unsigned short* __restrict__ Ybf) {
    int bat = blockIdx.y;
    int dl = threadIdx.x & 31;
    int s  = threadIdx.x >> 5;
    int d  = blockIdx.x * 32 + dl;
    const float* Xb = X + (long)bat * NN * DD;
    const float* Rb = R ? R + (long)bat * NN * DD : nullptr;
    float sum = 0.f, sq = 0.f;
    for (int n = s; n < NN; n += 8) {
        float v = Xb[n * DD + d];
        if (Rb) v += Rb[n * DD + d];
        sum += v; sq += v * v;
    }
    __shared__ float sm[8][32], s2[8][32];
    __shared__ float meanS[32], invS[32], wS[32], bS[32];
    sm[s][dl] = sum; s2[s][dl] = sq;
    __syncthreads();
    if (s == 0) {
        float ts = 0.f, tq = 0.f;
        for (int i = 0; i < 8; ++i) { ts += sm[i][dl]; tq += s2[i][dl]; }
        float m = ts * (1.0f / NN);
        float var = tq * (1.0f / NN) - m * m;
        meanS[dl] = m; invS[dl] = frsq(var + 1e-5f);
        wS[dl] = w[d]; bS[dl] = b[d];
    }
    __syncthreads();
    float m = meanS[dl], inv = invS[dl], ww = wS[dl], bb = bS[dl];
    for (int n = s; n < NN; n += 8) {
        float v = Xb[n * DD + d];
        if (Rb) v += Rb[n * DD + d];
        float y = (v - m) * inv * ww + bb;
        long idx = ((long)bat * NN + n) * DD + d;
        if (Yf)  Yf[idx] = y;
        if (Ybf) Ybf[idx] = f2bf(y);
    }
}

// ---------------------------------------------------------------------------
// adapt_bias (folded fusion) + softmax over m + exp -> eab (bf16)
// grid = B*N blocks (one output row each), block 256 (2 m per thread)
// ---------------------------------------------------------------------------
__global__ void adapt_eab_kernel(const float* __restrict__ cost, const float* __restrict__ coords,
                                 const float* __restrict__ dw1, const float* __restrict__ db1,
                                 const float* __restrict__ aw1, const float* __restrict__ ab1,
                                 const float* __restrict__ fold,
                                 const float* __restrict__ gate_b, const float* __restrict__ outlin_b,
                                 const float* __restrict__ alpha,
                                 unsigned short* __restrict__ eab) {
    int bn = blockIdx.x;            // 0..B*N-1
    int bat = bn >> 9, n = bn & (NN - 1);
    int t = threadIdx.x;
    __shared__ float sdw1[128], sdb1[128], su_d[128], sv_d[128];
    __shared__ float saw1[128], sab1[128], su_a[128], sv_a[128];
    if (t < 128) {
        sdw1[t] = dw1[t]; sdb1[t] = db1[t]; su_d[t] = fold[t];       sv_d[t] = fold[128 + t];
        saw1[t] = aw1[t]; sab1[t] = ab1[t]; su_a[t] = fold[256 + t]; sv_a[t] = fold[384 + t];
    }
    __syncthreads();
    float cd = fold[512], ed = fold[513], ca = fold[514], ea = fold[515];
    float gb = gate_b[0], ob = outlin_b[0], al = alpha[0];
    float cx = coords[(bat * NN + n) * 2 + 0], cy = coords[(bat * NN + n) * 2 + 1];
    float abv[2];
    #pragma unroll
    for (int i = 0; i < 2; ++i) {
        int m = t + i * 256;
        float c  = cost[(long)bn * NN + m];
        float dx = cx - coords[(bat * NN + m) * 2 + 0];
        float dy = cy - coords[(bat * NN + m) * 2 + 1];
        float ang = atan2f(dy, dx);
        float sd = cd, gd = ed, sa = ca, ga = ea;
        for (int j = 0; j < 128; ++j) {
            float hd = fmaxf(c * sdw1[j] + sdb1[j], 0.f);
            sd += hd * su_d[j]; gd += hd * sv_d[j];
            float ha = fmaxf(ang * saw1[j] + sab1[j], 0.f);
            sa += ha * su_a[j]; ga += ha * sv_a[j];
        }
        float g = bsig(gd + ga + gb);
        abv[i] = (g * sd + (1.f - g) * sa + ob) * al;
    }
    // softmax over 512, then exp, then bf16
    __shared__ float red[256];
    float mx = fmaxf(abv[0], abv[1]);
    red[t] = mx; __syncthreads();
    for (int off = 128; off > 0; off >>= 1) {
        if (t < off) red[t] = fmaxf(red[t], red[t + off]);
        __syncthreads();
    }
    mx = red[0]; __syncthreads();
    float e0 = __expf(abv[0] - mx), e1 = __expf(abv[1] - mx);
    red[t] = e0 + e1; __syncthreads();
    for (int off = 128; off > 0; off >>= 1) {
        if (t < off) red[t] += red[t + off];
        __syncthreads();
    }
    float inv = frcp(red[0]);
    eab[(long)bn * NN + t]       = f2bf(__expf(e0 * inv));
    eab[(long)bn * NN + t + 256] = f2bf(__expf(e1 * inv));
}

// ---------------------------------------------------------------------------
// K softmax over tokens (axis=1) per (b,h), then build transposed bf16
// [eK*V | eK] (O=256, K=N) for the batched AFT aggregation GEMM.
// grid = B*D blocks, block 256 (2 tokens each)
// ---------------------------------------------------------------------------
__global__ void ksoftmax_kernel(const float* __restrict__ Klog, const float* __restrict__ V,
                                unsigned short* __restrict__ EKVt) {
    int bat = blockIdx.x >> 7, h = blockIdx.x & 127;
    int t = threadIdx.x;
    const float* Kb = Klog + (long)bat * NN * DD;
    float v0 = Kb[t * DD + h], v1 = Kb[(t + 256) * DD + h];
    __shared__ float red[256];
    red[t] = fmaxf(v0, v1); __syncthreads();
    for (int off = 128; off > 0; off >>= 1) {
        if (t < off) red[t] = fmaxf(red[t], red[t + off]);
        __syncthreads();
    }
    float mx = red[0]; __syncthreads();
    float e0 = __expf(v0 - mx), e1 = __expf(v1 - mx);
    red[t] = e0 + e1; __syncthreads();
    for (int off = 128; off > 0; off >>= 1) {
        if (t < off) red[t] += red[t + off];
        __syncthreads();
    }
    float inv = frcp(red[0]);
    float ek0 = __expf(e0 * inv), ek1 = __expf(e1 * inv);
    const float* Vb = V + (long)bat * NN * DD;
    unsigned short* Eb = EKVt + (long)bat * 256 * NN;
    Eb[h * NN + t]               = f2bf(ek0 * Vb[t * DD + h]);
    Eb[h * NN + t + 256]         = f2bf(ek1 * Vb[(t + 256) * DD + h]);
    Eb[(128 + h) * NN + t]       = f2bf(ek0);
    Eb[(128 + h) * NN + t + 256] = f2bf(ek1);
}

// ---------------------------------------------------------------------------
// Yt input: bf16( Q * num / den )  from packed ND = [num | den]
// ---------------------------------------------------------------------------
__global__ void ytin_kernel(const float* __restrict__ Q, const float* __restrict__ ND,
                            unsigned short* __restrict__ out) {
    int idx = blockIdx.x * 256 + threadIdx.x;   // MM*DD
    int m = idx >> 7, h = idx & 127;
    int bat = m >> 9, n = m & (NN - 1);
    const float* nd = ND + ((long)bat * NN + n) * 256;
    out[idx] = f2bf(Q[idx] * nd[h] * frcp(nd[128 + h]));
}

// ---------------------------------------------------------------------------
// bf16 WMMA GEMM: C[m,o] = act( sum_k A[m,k]*Wt[o,k] + bias[o] )
// A: (M,K) bf16 row-major.  Wt: (O,K) bf16 row-major (pre-transposed weight).
// 16(M)x64(O) tile per wave: 4 f32 accumulators share one A fragment.
// Double-buffered fragments (a[2], b[2][4]) so k-step i+1's 10 b128 loads
// issue while step i's 4 v_wmma execute. 4 waves/block, K fully unrolled.
// grid.z = batch with element strides sA/sW/sC.
// VGPR layouts per CDNA5 ISA 7.12.2:
//   A(16x32 bf16): lane m=L%16, VGPR r holds K = (r>=4?16:0)+half*8+2(r&3) {,+1}
//   B(32x16 bf16): lane n=L%16, VGPR r holds K = (r>=4?16:0)+half*8+2(r&3) {,+1}
//   C(16x16 f32):  VGPR r -> row m0+half*8+r, col o0+(L%16)
// ---------------------------------------------------------------------------
template <int K>
__global__ __launch_bounds__(128)
void gemm_bf16_kernel(const unsigned short* __restrict__ A,
                      const unsigned short* __restrict__ Wt,
                      const float* __restrict__ bias,
                      float* __restrict__ Cf, unsigned short* __restrict__ Cbf,
                      int O, int act, long sA, long sW, long sC) {
    const int wave = threadIdx.x >> 5;
    const int lane = threadIdx.x & 31;
    const int tile = blockIdx.x * 4 + wave;
    const int tilesO = O >> 6;                 // 64-wide output groups
    const int m0  = (tile / tilesO) << 4;
    const int og0 = (tile % tilesO) << 6;
    const int half = lane >> 4;
    const int ln = lane & 15;

    const unsigned short* arow = A + (long)blockIdx.z * sA + (long)(m0 + ln) * K;
    const unsigned short* brow[4];
    #pragma unroll
    for (int tl = 0; tl < 4; ++tl)
        brow[tl] = Wt + (long)blockIdx.z * sW + (long)(og0 + tl * 16 + ln) * K;

    union Frag { unsigned int u[8]; v16bf v; };
    Frag a[2], b[2][4];

    auto loadA = [&](Frag& f, int k0) {
        #pragma unroll
        for (int r = 0; r < 8; ++r) {
            const int ka = k0 + ((r >= 4) ? 16 : 0) + half * 8 + ((r & 3) << 1);
            f.u[r] = *(const unsigned int*)(arow + ka);
        }
    };
    auto loadB = [&](Frag& f, const unsigned short* base, int k0) {
        #pragma unroll
        for (int r = 0; r < 8; ++r)
            f.u[r] = *(const unsigned int*)(base + k0 + ((r >= 4) ? 16 : 0) + half * 8 + ((r & 3) << 1));
    };

    loadA(a[0], 0);
    #pragma unroll
    for (int tl = 0; tl < 4; ++tl) loadB(b[0][tl], brow[tl], 0);

    v8f acc[4] = {};
    #pragma unroll
    for (int k0 = 0; k0 < K; k0 += 32) {
        const int cur = (k0 >> 5) & 1;
        const int nxt = cur ^ 1;
        if (k0 + 32 < K) {                       // prefetch next k-step
            loadA(a[nxt], k0 + 32);
            #pragma unroll
            for (int tl = 0; tl < 4; ++tl) loadB(b[nxt][tl], brow[tl], k0 + 32);
        }
        #pragma unroll
        for (int tl = 0; tl < 4; ++tl)
            acc[tl] = __builtin_amdgcn_wmma_f32_16x16x32_bf16(false, a[cur].v, false, b[cur][tl].v,
                                                              (short)0, acc[tl], false, false);
    }

    #pragma unroll
    for (int tl = 0; tl < 4; ++tl) {
        const int col = og0 + tl * 16 + ln;
        const float bi = bias ? bias[col] : 0.0f;
        #pragma unroll
        for (int r = 0; r < 8; ++r) {
            const int row = m0 + half * 8 + r;
            float v = acc[tl][r] + bi;
            if (act == 1)      v = fmaxf(v, 0.0f);
            else if (act == 2) v = bsig(v);
            const long idx = (long)blockIdx.z * sC + (long)row * O + col;
            if (Cf)  Cf[idx] = v;
            if (Cbf) Cbf[idx] = f2bf(v);
        }
    }
}

// ---------------------------------------------------------------------------
// Workspace layout (bytes)
// ---------------------------------------------------------------------------
static constexpr size_t SZ_MD_F  = (size_t)MM * DD * 4;   // 512 KB
static constexpr size_t SZ_MD_BF = (size_t)MM * DD * 2;   // 256 KB
static constexpr size_t OFF_RN   = 0;
static constexpr size_t OFF_CN   = OFF_RN   + SZ_MD_F;
static constexpr size_t OFF_Q    = OFF_CN   + SZ_MD_F;
static constexpr size_t OFF_KLOG = OFF_Q    + SZ_MD_F;
static constexpr size_t OFF_V    = OFF_KLOG + SZ_MD_F;
static constexpr size_t OFF_ND   = OFF_V    + SZ_MD_F;
static constexpr size_t OFF_MHC  = OFF_ND   + (size_t)MM * 256 * 4;
static constexpr size_t OFF_MH   = OFF_MHC  + SZ_MD_F;
static constexpr size_t OFF_X    = OFF_MH   + SZ_MD_F;
static constexpr size_t OFF_FF2  = OFF_X    + SZ_MD_F;
static constexpr size_t OFF_FOLD = OFF_FF2  + SZ_MD_F;
static constexpr size_t OFF_RNBF = OFF_FOLD + 4096;
static constexpr size_t OFF_CNBF = OFF_RNBF + SZ_MD_BF;
static constexpr size_t OFF_EAB  = OFF_CNBF + SZ_MD_BF;
static constexpr size_t OFF_EKVT = OFF_EAB  + (size_t)BB * NN * NN * 2;
static constexpr size_t OFF_YTIN = OFF_EKVT + (size_t)BB * 256 * NN * 2;
static constexpr size_t OFF_YTBF = OFF_YTIN + SZ_MD_BF;
static constexpr size_t OFF_XBF  = OFF_YTBF + SZ_MD_BF;
static constexpr size_t OFF_F1BF = OFF_XBF  + SZ_MD_BF;
static constexpr size_t OFF_WQ   = OFF_F1BF + (size_t)MM * FH * 2;
static constexpr size_t OFF_WK   = OFF_WQ   + (size_t)DD * DD * 2;
static constexpr size_t OFF_WV   = OFF_WK   + (size_t)DD * DD * 2;
static constexpr size_t OFF_WP   = OFF_WV   + (size_t)DD * DD * 2;
static constexpr size_t OFF_WMHC = OFF_WP   + (size_t)DD * DD * 2;
static constexpr size_t OFF_WF1  = OFF_WMHC + (size_t)DD * DD * 2;
static constexpr size_t OFF_WF2  = OFF_WF1  + (size_t)DD * FH * 2;
// total ~= 9.9 MB

extern "C" void kernel_launch(void* const* d_in, const int* in_sizes, int n_in,
                              void* d_out, int out_size, void* d_ws, size_t ws_size,
                              hipStream_t stream) {
    const float* row_emb = (const float*)d_in[0];
    const float* col_emb = (const float*)d_in[1];
    const float* cost    = (const float*)d_in[2];
    const float* coords  = (const float*)d_in[3];
    const float* alpha   = (const float*)d_in[4];
    const float* n1_w = (const float*)d_in[5],  *n1_b = (const float*)d_in[6];
    const float* n2_w = (const float*)d_in[7],  *n2_b = (const float*)d_in[8];
    const float* n3_w = (const float*)d_in[9],  *n3_b = (const float*)d_in[10];
    const float* fn1_w = (const float*)d_in[11], *fn1_b = (const float*)d_in[12];
    const float* fn2_w = (const float*)d_in[13], *fn2_b = (const float*)d_in[14];
    const float* q_w = (const float*)d_in[15],  *q_b = (const float*)d_in[16];
    const float* k_w = (const float*)d_in[17],  *k_b = (const float*)d_in[18];
    const float* v_w = (const float*)d_in[19],  *v_b = (const float*)d_in[20];
    const float* p_w = (const float*)d_in[21],  *p_b = (const float*)d_in[22];
    const float* mhc_w = (const float*)d_in[23], *mhc_b = (const float*)d_in[24];
    const float* dist_w1 = (const float*)d_in[25], *dist_b1 = (const float*)d_in[26];
    const float* dist_w2 = (const float*)d_in[27], *dist_b2 = (const float*)d_in[28];
    const float* ang_w1 = (const float*)d_in[29], *ang_b1 = (const float*)d_in[30];
    const float* ang_w2 = (const float*)d_in[31], *ang_b2 = (const float*)d_in[32];
    const float* gate_w = (const float*)d_in[33], *gate_b = (const float*)d_in[34];
    const float* outlin_w = (const float*)d_in[35], *outlin_b = (const float*)d_in[36];
    const float* ffn_w1 = (const float*)d_in[37], *ffn_b1 = (const float*)d_in[38];
    const float* ffn_w2 = (const float*)d_in[39], *ffn_b2 = (const float*)d_in[40];
    (void)in_sizes; (void)n_in; (void)out_size; (void)ws_size;

    char* ws = (char*)d_ws;
    auto F = [&](size_t off) { return (float*)(ws + off); };
    auto U = [&](size_t off) { return (unsigned short*)(ws + off); };

    // 1) pack weights -> (O,K) bf16
    packwt_kernel<<<64, 256, 0, stream>>>(q_w,   U(OFF_WQ),   DD, DD);
    packwt_kernel<<<64, 256, 0, stream>>>(k_w,   U(OFF_WK),   DD, DD);
    packwt_kernel<<<64, 256, 0, stream>>>(v_w,   U(OFF_WV),   DD, DD);
    packwt_kernel<<<64, 256, 0, stream>>>(p_w,   U(OFF_WP),   DD, DD);
    packwt_kernel<<<64, 256, 0, stream>>>(mhc_w, U(OFF_WMHC), DD, DD);
    packwt_kernel<<<256, 256, 0, stream>>>(ffn_w1, U(OFF_WF1), DD, FH);
    packwt_kernel<<<256, 256, 0, stream>>>(ffn_w2, U(OFF_WF2), FH, DD);

    // 2) fold fusion vectors
    fold_kernel<<<1, 128, 0, stream>>>(dist_w2, dist_b2, ang_w2, ang_b2,
                                       outlin_w, gate_w, F(OFF_FOLD));

    // 3) instance norms of inputs
    inorm_kernel<<<dim3(DD / 32, BB), 256, 0, stream>>>(row_emb, nullptr, n1_w, n1_b,
                                                        F(OFF_RN), U(OFF_RNBF));
    inorm_kernel<<<dim3(DD / 32, BB), 256, 0, stream>>>(col_emb, nullptr, n2_w, n2_b,
                                                        F(OFF_CN), U(OFF_CNBF));

    // 4) adapt_bias -> softmax -> exp -> eab (bf16)
    adapt_eab_kernel<<<MM, 256, 0, stream>>>(cost, coords, dist_w1, dist_b1, ang_w1, ang_b1,
                                             F(OFF_FOLD), gate_b, outlin_b, alpha, U(OFF_EAB));

    // 5) Q = sigmoid(rn@q_w+b), Klog = cn@k_w+b, V = cn@v_w+b  (WMMA)
    gemm_bf16_kernel<DD><<<dim3(32, 1, 1), 128, 0, stream>>>(
        U(OFF_RNBF), U(OFF_WQ), q_b, F(OFF_Q), nullptr, DD, 2, 0, 0, 0);
    gemm_bf16_kernel<DD><<<dim3(32, 1, 1), 128, 0, stream>>>(
        U(OFF_CNBF), U(OFF_WK), k_b, F(OFF_KLOG), nullptr, DD, 0, 0, 0, 0);
    gemm_bf16_kernel<DD><<<dim3(32, 1, 1), 128, 0, stream>>>(
        U(OFF_CNBF), U(OFF_WV), v_b, F(OFF_V), nullptr, DD, 0, 0, 0, 0);

    // 6) token softmax of K, build [eK*V | eK]^T bf16
    ksoftmax_kernel<<<BB * DD, 256, 0, stream>>>(F(OFF_KLOG), F(OFF_V), U(OFF_EKVT));

    // 7) batched ND = eab @ [eK*V | eK]   (M=512, K=512, O=256, z=B)
    gemm_bf16_kernel<NN><<<dim3(32, 1, BB), 128, 0, stream>>>(
        U(OFF_EAB), U(OFF_EKVT), nullptr, F(OFF_ND), nullptr, 256, 0,
        (long)NN * NN, (long)256 * NN, (long)NN * 256);

    // 8) Yt input = bf16(Q * num / den)
    ytin_kernel<<<MM * DD / 256, 256, 0, stream>>>(F(OFF_Q), F(OFF_ND), U(OFF_YTIN));

    // 9) Yt = ytin @ p_w + p_b (bf16 out), then mhc GEMM (f32 out)
    gemm_bf16_kernel<DD><<<dim3(32, 1, 1), 128, 0, stream>>>(
        U(OFF_YTIN), U(OFF_WP), p_b, nullptr, U(OFF_YTBF), DD, 0, 0, 0, 0);
    gemm_bf16_kernel<DD><<<dim3(32, 1, 1), 128, 0, stream>>>(
        U(OFF_YTBF), U(OFF_WMHC), mhc_b, F(OFF_MHC), nullptr, DD, 0, 0, 0, 0);

    // 10) mh = inorm(mhc_out), x = inorm(rn + mh)
    inorm_kernel<<<dim3(DD / 32, BB), 256, 0, stream>>>(F(OFF_MHC), nullptr, n3_w, n3_b,
                                                        F(OFF_MH), nullptr);
    inorm_kernel<<<dim3(DD / 32, BB), 256, 0, stream>>>(F(OFF_RN), F(OFF_MH), fn1_w, fn1_b,
                                                        F(OFF_X), U(OFF_XBF));

    // 11) FFN: relu(x@W1+b1) (bf16) then @W2+b2 (f32)
    gemm_bf16_kernel<DD><<<dim3(128, 1, 1), 128, 0, stream>>>(
        U(OFF_XBF), U(OFF_WF1), ffn_b1, nullptr, U(OFF_F1BF), FH, 1, 0, 0, 0);
    gemm_bf16_kernel<FH><<<dim3(32, 1, 1), 128, 0, stream>>>(
        U(OFF_F1BF), U(OFF_WF2), ffn_b2, F(OFF_FF2), nullptr, DD, 0, 0, 0, 0);

    // 12) out = inorm(x + ff, fn2) -> d_out
    inorm_kernel<<<dim3(DD / 32, BB), 256, 0, stream>>>(F(OFF_X), F(OFF_FF2), fn2_w, fn2_b,
                                                        (float*)d_out, nullptr);
}
